// MNISTCanny_44126493999425
// MI455X (gfx1250) — matrix operand order repaired and checked by reading, not compiled, not run on hardware
//
#include <hip/hip_runtime.h>
#include <hip/hip_bf16.h>

typedef __attribute__((ext_vector_type(16))) _Float16 v16h;
typedef __attribute__((ext_vector_type(8)))  _Float16 v8h;
typedef __attribute__((ext_vector_type(8)))  float    v8f;

#define EPS_C    1e-9f
#define GAMMA_C  0.005f
#define HIGH_T_C 0.02f
#define LOW_T_C  0.01f

__device__ __forceinline__ int clamp31(int v) { return v < 0 ? 0 : (v > 31 ? 31 : v); }

__global__ __launch_bounds__(256)
void MNISTCanny_44126493999425_kernel(const float* __restrict__ x,
                                      const float* __restrict__ mask,
                                      const float* __restrict__ g25,
                                      const float* __restrict__ smaj,
                                      const float* __restrict__ smin,
                                      float* __restrict__ out) {
    // ---- LDS working set (~41 KB, one WGP) ----
    __shared__ __align__(16) _Float16 Gh[32][32];   // gaussian band matrix (symmetric: G == G^T)
    __shared__ __align__(16) _Float16 XhT[32][32];  // input image, TRANSPOSED (B-operand rows)
    __shared__ __align__(16) _Float16 MhT[32][32];  // mask, TRANSPOSED (B-operand rows)
    __shared__ __align__(16) _Float16 T1h[32][32];  // G @ X   (row-major, A-operand for stage 2)
    __shared__ __align__(16) _Float16 T2h[32][32];  // G @ M
    __shared__ float Mf[32][32];       // mask f32 (for erosion)
    __shared__ float xsn[32][32];      // (G@X)@G
    __shared__ float bld[32][32];      // (G@M)@G  (bleed)
    __shared__ float xsf[32][32];      // xs = xsn / (bld + 1e-12)
    __shared__ float djs[32][32];      // W-derivative intermediate
    __shared__ float dis[32][32];      // H-derivative intermediate
    __shared__ float magL[32][32];     // gradient magnitude
    __shared__ unsigned char lowm[1024], bufA[1024], bufB[1024];
    __shared__ int s_changed, s_maxbits;

    const int tid  = threadIdx.x;
    const int wave = tid >> 5;
    const int lane = tid & 31;
    const int Ml   = lane & 15;     // M (A/D) or N (B/D) sub-index
    const int half = lane >> 4;

    // ---- Phase 0: stage inputs into LDS ----
    if (tid == 0) { s_changed = 0; s_maxbits = 0; }
    #pragma unroll
    for (int q = 0; q < 4; ++q) {
        int p = tid + q * 256, i = p >> 5, j = p & 31;
        float xv = x[p], mv = mask[p];
        XhT[j][i] = (_Float16)xv;      // transposed: row n = column n of X
        Mf[i][j]  = mv;
        MhT[j][i] = (_Float16)mv;
        int d = j - i + 12;
        Gh[i][j] = (d >= 0 && d < 25) ? (_Float16)g25[d] : (_Float16)0.0f;
    }
    __syncthreads();

    // fragment builders: A row run = {8h..8h+7} ++ {16+8h..16+8h+7}; B row run = {8h..8h+7} ++ {16+8h..16+8h+7}
    auto loadA = [&](const _Float16* row) -> v16h {
        v8h a0 = *(const v8h*)(row + half * 8);
        v8h a1 = *(const v8h*)(row + 16 + half * 8);
        return __builtin_shufflevector(a0, a1, 0,1,2,3,4,5,6,7,8,9,10,11,12,13,14,15);
    };
    auto loadB = [&](const _Float16* rowT) -> v16h {
        v8h b0 = *(const v8h*)(rowT + half * 8);
        v8h b1 = *(const v8h*)(rowT + 16 + half * 8);
        return __builtin_shufflevector(b0, b1, 0,1,2,3,4,5,6,7,8,9,10,11,12,13,14,15);
    };

    const int t  = wave & 3, src = wave >> 2;
    const int tm = (t >> 1) * 16, tn = (t & 1) * 16;

    // ---- Phase 1: WMMA  T1 = G@X (waves 0-3), T2 = G@M (waves 4-7) ----
    {
        const _Float16 (*BT)[32] = src ? MhT : XhT;       // B^T rows = B columns
        v16h a = loadA(&Gh[tm + Ml][0]);
        v16h b = loadB(&BT[tn + Ml][0]);
        v8f c = {};
        c = __builtin_amdgcn_wmma_f32_16x16x32_f16(false, a, false, b, (short)0, c, false, false);
        #pragma unroll
        for (int r = 0; r < 8; ++r) {
            int Mo = tm + r + 8 * half;
            if (src) T2h[Mo][tn + Ml] = (_Float16)c[r];
            else     T1h[Mo][tn + Ml] = (_Float16)c[r];
        }
    }
    __syncthreads();

    // ---- Phase 2: WMMA  xsn = T1@G, bld = T2@G  (G symmetric -> B rows = G rows) ----
    {
        const _Float16 (*Ap)[32] = src ? T2h : T1h;
        v16h a = loadA(&Ap[tm + Ml][0]);
        v16h b = loadB(&Gh[tn + Ml][0]);                  // G[K][N] == G[N][K]
        v8f c = {};
        c = __builtin_amdgcn_wmma_f32_16x16x32_f16(false, a, false, b, (short)0, c, false, false);
        #pragma unroll
        for (int r = 0; r < 8; ++r) {
            int Mo = tm + r + 8 * half;
            if (src) bld[Mo][tn + Ml] = c[r];
            else     xsn[Mo][tn + Ml] = c[r];
        }
    }
    __syncthreads();

    // ---- Phase 3: xs = num / (bleed + 1e-12) ----
    #pragma unroll
    for (int q = 0; q < 4; ++q) {
        int p = tid + q * 256, i = p >> 5, j = p & 31;
        xsf[i][j] = xsn[i][j] / (bld[i][j] + 1e-12f);
    }
    __syncthreads();

    // ---- Phase 4: first-pass sobel (edge-clamped) ----
    const float sm0 = smaj[0], sm1 = smaj[1], sm2 = smaj[2];
    const float sn0 = smin[0], sn1 = smin[1], sn2 = smin[2];
    #pragma unroll
    for (int q = 0; q < 4; ++q) {
        int p = tid + q * 256, i = p >> 5, j = p & 31;
        djs[i][j] = sm0 * xsf[i][clamp31(j - 1)] + sm1 * xsf[i][j] + sm2 * xsf[i][clamp31(j + 1)];
        dis[i][j] = sm0 * xsf[clamp31(i - 1)][j] + sm1 * xsf[i][j] + sm2 * xsf[clamp31(i + 1)][j];
    }
    __syncthreads();

    // ---- Phase 5: magnitude ----
    #pragma unroll
    for (int q = 0; q < 4; ++q) {
        int p = tid + q * 256, i = p >> 5, j = p & 31;
        float js = sn0 * djs[clamp31(i - 1)][j] + sn1 * djs[i][j] + sn2 * djs[clamp31(i + 1)][j];
        float is = sn0 * dis[i][clamp31(j - 1)] + sn1 * dis[i][j] + sn2 * dis[i][clamp31(j + 1)];
        magL[i][j] = sqrtf(is * is + js * js + EPS_C);
    }
    __syncthreads();

    // ---- Phase 6: NMS in 4 directions + lm + hysteresis seeds ----
    #pragma unroll
    for (int q = 0; q < 4; ++q) {
        int p = tid + q * 256, i = p >> 5, j = p & 31;
        float js = sn0 * djs[clamp31(i - 1)][j] + sn1 * djs[i][j] + sn2 * djs[clamp31(i + 1)][j];
        float is = sn0 * dis[i][clamp31(j - 1)] + sn1 * dis[i][j] + sn2 * dis[i][clamp31(j + 1)];
        float ai = fabsf(is), aj = fabsf(js);
        float mag2 = is * is + js * js;
        float magv = magL[i][j];

        // eroded mask (zero border) & mag2>0
        bool er = (mag2 > 0.0f);
        #pragma unroll
        for (int di = -1; di <= 1; ++di)
            #pragma unroll
            for (int dj = -1; dj <= 1; ++dj) {
                int ii = i + di, jj = j + dj;
                bool inb = (ii >= 0) && (ii < 32) && (jj >= 0) && (jj < 32);
                er = er && inb && (Mf[ii & 31][jj & 31] != 0.0f);
            }

        bool ip0 = is >= 0.0f, jp0 = js >= 0.0f, im0 = is <= 0.0f, jm0 = js <= 0.0f;
        bool diag1 = (ip0 && jp0) || (im0 && jm0);
        bool diag2 = (im0 && jp0) || (ip0 && jm0);
        bool pts1 = diag1 && (ai >= aj) && er;
        bool pts2 = diag1 && (ai <= aj) && er;
        bool pts3 = diag2 && (ai <= aj) && er;
        bool pts4 = diag2 && (ai >= aj) && er;
        float w1 = aj / (ai + EPS_C);
        float w2 = ai / (pts2 ? aj : 1.0f);
        float w3 = ai / (pts3 ? aj : 1.0f);
        float w4 = aj / (pts4 ? ai : 1.0f);

        // wrap-shift magnitude (matches jnp.roll)
        auto magS = [&](int di, int dj) -> float {
            return magL[(i + di) & 31][(j + dj) & 31];
        };
        auto nms = [&](bool pts, float w,
                       int o1i, int o1j, int o2i, int o2j,
                       int m1i, int m1j, int m2i, int m2j,
                       bool bug, float& s, bool& lmax) {
            float ipv = magS(o2i, o2j) * w + magS(o1i, o1j) * (1.0f - w);
            float imv = magS(m2i, m2j) * w + magS(m1i, m1j) * (1.0f - w);
            lmax = (ipv <= magv) && (imv <= magv);
            float a1 = fmaxf(0.0f, -magv + GAMMA_C + ipv);
            float a2 = fmaxf(0.0f, -magv + GAMMA_C + (bug ? ipv : imv));
            s = pts ? fmaxf(a1, a2) : 0.0f;
        };

        float s1, s2, s3, s4; bool l1, l2, l3, l4;
        nms(pts1, w1,  1, 0,  1, 1,  -1, 0,  -1,-1, false, s1, l1);
        nms(pts2, w2,  0, 1,  1, 1,   0,-1,  -1,-1, false, s2, l2);
        nms(pts3, w3,  0, 1, -1, 1,   0,-1,   1,-1, true,  s3, l3);  // reference's c2a/c1a reuse bug
        nms(pts4, w4, -1, 0, -1, 1,   1, 0,   1,-1, false, s4, l4);

        bool lm = false;
        if (pts1) lm = l1;
        if (pts2) lm = l2;
        if (pts3) lm = l3;
        if (pts4) lm = l4;

        out[p]        = s1;
        out[1024 + p] = s2;
        out[2048 + p] = s3;
        out[3072 + p] = s4;
        out[4096 + p] = lm ? 1.0f : 0.0f;

        lowm[p] = (lm && magv >= LOW_T_C) ? 1 : 0;
        bufA[p] = (lm && magv >= HIGH_T_C) ? 1 : 0;
    }
    __syncthreads();

    // ---- Phase 7: hysteresis flood fill (8-connectivity, fixed point) ----
    unsigned char* cur = bufA;
    unsigned char* nxt = bufB;
    for (int it = 0; it < 2048; ++it) {
        #pragma unroll
        for (int q = 0; q < 4; ++q) {
            int p = tid + q * 256, i = p >> 5, j = p & 31;
            unsigned char nv = 0;
            if (lowm[p]) {
                bool any = false;
                #pragma unroll
                for (int di = -1; di <= 1; ++di)
                    #pragma unroll
                    for (int dj = -1; dj <= 1; ++dj) {
                        int ii = i + di, jj = j + dj;
                        if (ii >= 0 && ii < 32 && jj >= 0 && jj < 32 && cur[ii * 32 + jj]) any = true;
                    }
                nv = any ? 1 : 0;
            }
            nxt[p] = nv;
            if (nv != cur[p]) s_changed = 1;   // benign race, all write 1
        }
        __syncthreads();
        int done = (s_changed == 0);
        __syncthreads();                        // everyone read flag before reset
        if (tid == 0) s_changed = 0;
        unsigned char* tswap = cur; cur = nxt; nxt = tswap;
        __syncthreads();                        // reset visible before next writes
        if (done) break;
    }

    // ---- Phase 8: normalize by global max ----
    #pragma unroll
    for (int q = 0; q < 4; ++q) {
        int p = tid + q * 256, i = p >> 5, j = p & 31;
        float mm = magL[i][j] * (cur[p] ? 1.0f : 0.0f);
        atomicMax(&s_maxbits, __float_as_int(mm));  // mm >= 0 -> int order == float order
    }
    __syncthreads();
    float mx  = __int_as_float(s_maxbits);
    float den = mx > 0.0f ? mx : 1.0f;
    #pragma unroll
    for (int q = 0; q < 4; ++q) {
        int p = tid + q * 256, i = p >> 5, j = p & 31;
        out[5120 + p] = (magL[i][j] * (cur[p] ? 1.0f : 0.0f)) / den;
    }
}

extern "C" void kernel_launch(void* const* d_in, const int* in_sizes, int n_in,
                              void* d_out, int out_size, void* d_ws, size_t ws_size,
                              hipStream_t stream) {
    (void)in_sizes; (void)n_in; (void)out_size; (void)d_ws; (void)ws_size;
    const float* x    = (const float*)d_in[0];
    const float* mask = (const float*)d_in[1];
    const float* g25  = (const float*)d_in[2];
    const float* smaj = (const float*)d_in[3];
    const float* smin = (const float*)d_in[4];
    float* out = (float*)d_out;
    MNISTCanny_44126493999425_kernel<<<dim3(1), dim3(256), 0, stream>>>(x, mask, g25, smaj, smin, out);
}
